// CrossAttention_44538810859827
// MI455X (gfx1250) — matrix operand order repaired and checked by reading, not compiled, not run on hardware
//
#include <hip/hip_runtime.h>

#define WIN_TOKENS 128
#define HEADS 8
#define HEAD_DIM 32
#define DIM 256
#define HIDDEN 512
#define NWIN 512
#define QK_SCALE 0.17677669529663687f

typedef __attribute__((ext_vector_type(16))) __bf16 v16bf;
typedef __attribute__((ext_vector_type(8))) float v8f;
typedef int v4i_vec __attribute__((vector_size(16)));

union FragBF { v16bf v; unsigned u[8]; };

__device__ __forceinline__ v8f wmma_bf16(v16bf a, v16bf b, v8f c) {
  return __builtin_amdgcn_wmma_f32_16x16x32_bf16(false, a, false, b, (short)0, c,
                                                 false, false);
}

__device__ __forceinline__ unsigned short f2bf(float f) {
  union { float f; unsigned u; } x; x.f = f;
  unsigned r = (x.u + 0x7FFFu + ((x.u >> 16) & 1u)) >> 16;  // RNE
  return (unsigned short)r;
}
__device__ __forceinline__ unsigned pack2bf(float a, float b) {
  return (unsigned)f2bf(a) | ((unsigned)f2bf(b) << 16);
}

// ---- CDNA5 async global->LDS (ASYNCcnt) with portable fallback ----
#if defined(__has_builtin)
#if __has_builtin(__builtin_amdgcn_global_load_async_to_lds_b128) && \
    __has_builtin(__builtin_amdgcn_s_wait_asynccnt)
#define HAVE_ASYNC_LDS 1
#endif
#endif
#ifndef HAVE_ASYNC_LDS
#define HAVE_ASYNC_LDS 0
#endif

__device__ __forceinline__ void copy16_to_lds(unsigned short* ldst,
                                              const unsigned short* gsrc) {
#if HAVE_ASYNC_LDS
  __builtin_amdgcn_global_load_async_to_lds_b128(
      (v4i_vec*)(void*)const_cast<unsigned short*>(gsrc),
      (v4i_vec*)(void*)ldst, 0, 0);
#else
  *(uint4*)ldst = *(const uint4*)gsrc;
#endif
}
__device__ __forceinline__ void copy_wait() {
#if HAVE_ASYNC_LDS
  __builtin_amdgcn_s_wait_asynccnt(0);
#endif
}

// A fragment (16x32 bf16) from row-major ushort matrix, rows Mbase.., cols Kbase..
// Contiguous k layout: lane<16 -> M=lane, K pairs {0..7} in v0..3, {8..15} in v4..7
//             lane>=16 -> M=lane-16, K {16..23},{24..31}
__device__ __forceinline__ v16bf load_A(const unsigned short* m, int Mbase,
                                        int Kbase, int stride) {
  const int lane = threadIdx.x & 31;
  const unsigned short* p = m + (Mbase + (lane & 15)) * stride;
  const int kb0 = Kbase + ((lane < 16) ? 0 : 16);
  const int kb1 = kb0 + 8;
  FragBF f;
#pragma unroll
  for (int j = 0; j < 4; ++j) {
    f.u[j]     = *(const unsigned*)(p + kb0 + 2 * j);
    f.u[4 + j] = *(const unsigned*)(p + kb1 + 2 * j);
  }
  return f.v;
}

// B fragment (32x16 bf16) where B = M^T, M row-major [N][K] (bf16, LDS or global).
// Contiguous k layout: lane<16 -> N=lane, K=0..15 (pairs 2v,2v+1); lane>=16 -> N=lane-16, K=16..31
__device__ __forceinline__ v16bf load_BT(const unsigned short* m, int Nbase,
                                         int Kbase, long stride) {
  const int lane = threadIdx.x & 31;
  const unsigned short* p =
      m + (long)(Nbase + (lane & 15)) * stride + Kbase + ((lane < 16) ? 0 : 16);
  FragBF f;
#pragma unroll
  for (int j = 0; j < 8; ++j) f.u[j] = *(const unsigned*)(p + 2 * j);
  return f.v;
}

// B fragment from row-major B itself [K][N] (AV case: V is [token][dim]).
__device__ __forceinline__ v16bf load_B(const unsigned short* m, int Kbase,
                                        int Nbase, int stride) {
  const int lane = threadIdx.x & 31;
  const int n = Nbase + (lane & 15);
  const int kb = Kbase + ((lane < 16) ? 0 : 16);
  FragBF f;
#pragma unroll
  for (int j = 0; j < 8; ++j) {
    unsigned lo = m[(kb + 2 * j) * stride + n];
    unsigned hi = m[(kb + 2 * j + 1) * stride + n];
    f.u[j] = lo | (hi << 16);
  }
  return f.v;
}

// ---------------- Kernel 0: one-shot f32 -> bf16 weight conversion ----------------
// segments: proj_w [0,65536), mw1 [65536,196608), mw2 [196608,327680)
__global__ __launch_bounds__(256) void cvt_w_kernel(
    const float* __restrict__ pw, const float* __restrict__ w1,
    const float* __restrict__ w2, unsigned short* __restrict__ out) {
  const int i0 = (blockIdx.x * 256 + threadIdx.x) * 8;
  const float* src; int rel;
  if (i0 < 65536)       { src = pw; rel = i0; }
  else if (i0 < 196608) { src = w1; rel = i0 - 65536; }
  else                  { src = w2; rel = i0 - 196608; }
#pragma unroll
  for (int j = 0; j < 8; j += 2)
    *(unsigned*)(out + i0 + j) = pack2bf(src[rel + j], src[rel + j + 1]);
}

// ---------------- Kernel 1: LayerNorm q/k/v -> bf16 window/head layout ----------------
__global__ __launch_bounds__(256) void ln_qkv_kernel(
    const float* __restrict__ q, const float* __restrict__ k,
    const float* __restrict__ v, const float* __restrict__ gq,
    const float* __restrict__ bq, const float* __restrict__ gk,
    const float* __restrict__ bk, const float* __restrict__ gv,
    const float* __restrict__ bv, unsigned short* __restrict__ qh,
    unsigned short* __restrict__ kh, unsigned short* __restrict__ vh) {
  const int wv = threadIdx.x >> 5, lane = threadIdx.x & 31;
  const long r = (long)blockIdx.x * 8 + wv;  // 0 .. 3*65536-1
  const int tensor = (int)(r >> 16);
  const int g = (int)(r & 65535);
  const float* src; const float* gam; const float* bet; unsigned short* dst;
  if (tensor == 0)      { src = q; gam = gq; bet = bq; dst = qh; }
  else if (tensor == 1) { src = k; gam = gk; bet = bk; dst = kh; }
  else                  { src = v; gam = gv; bet = bv; dst = vh; }
  const float* row = src + (long)g * DIM;
  const int d0 = lane * 8;
  float x[8]; float s = 0.f, ss = 0.f;
#pragma unroll
  for (int j = 0; j < 8; ++j) { x[j] = row[d0 + j]; s += x[j]; ss += x[j] * x[j]; }
#pragma unroll
  for (int d = 16; d >= 1; d >>= 1) {
    s += __shfl_xor(s, d, 32);
    ss += __shfl_xor(ss, d, 32);
  }
  const float mean = s * (1.f / DIM);
  const float rstd = rsqrtf(ss * (1.f / DIM) - mean * mean + 1e-5f);
  // window partition mapping: g = ((b*8+t)*64+h)*64+w
  const int w = g & 63, h = (g >> 6) & 63, t = (g >> 12) & 7, b = g >> 15;
  const int win = ((b * 4 + (t >> 1)) * 8 + (h >> 3)) * 8 + (w >> 3);
  const int n = ((t & 1) * 8 + (h & 7)) * 8 + (w & 7);
  unsigned short* dp = dst + ((((long)win * HEADS + (d0 >> 5)) * WIN_TOKENS + n) *
                                  HEAD_DIM + (d0 & 31));
#pragma unroll
  for (int j = 0; j < 8; j += 2) {
    float a = (x[j] - mean) * rstd * gam[d0 + j] + bet[d0 + j];
    float bb = (x[j + 1] - mean) * rstd * gam[d0 + j + 1] + bet[d0 + j + 1];
    *(unsigned*)(dp + j) = pack2bf(a, bb);
  }
}

// ---------------- Kernel 2: windowed attention, one block per (window, head) ----------------
__global__ __launch_bounds__(256) void win_attn_kernel(
    const unsigned short* __restrict__ qh, const unsigned short* __restrict__ kh,
    const unsigned short* __restrict__ vh, const float* __restrict__ bias_table,
    unsigned short* __restrict__ ao) {
  __shared__ __attribute__((aligned(16))) unsigned short sQ[WIN_TOKENS * HEAD_DIM];
  __shared__ __attribute__((aligned(16))) unsigned short sK[WIN_TOKENS * HEAD_DIM];
  __shared__ __attribute__((aligned(16))) unsigned short sV[WIN_TOKENS * HEAD_DIM];
  __shared__ __attribute__((aligned(16))) unsigned short sP[WIN_TOKENS * WIN_TOKENS];
  __shared__ float sBias[675];

  const int tid = threadIdx.x;
  const int wv = tid >> 5, lane = tid & 31;
  const int win = blockIdx.x >> 3, head = blockIdx.x & 7;
  const long base = ((long)win * HEADS + head) * WIN_TOKENS * HEAD_DIM;

  {  // tile loads: 4096 bf16 each = 512 x b128, via async-to-LDS when available
    const unsigned short* q4 = qh + base;
    const unsigned short* k4 = kh + base;
    const unsigned short* v4 = vh + base;
    copy16_to_lds(sQ + tid * 8, q4 + tid * 8);
    copy16_to_lds(sQ + (tid + 256) * 8, q4 + (tid + 256) * 8);
    copy16_to_lds(sK + tid * 8, k4 + tid * 8);
    copy16_to_lds(sK + (tid + 256) * 8, k4 + (tid + 256) * 8);
    copy16_to_lds(sV + tid * 8, v4 + tid * 8);
    copy16_to_lds(sV + (tid + 256) * 8, v4 + (tid + 256) * 8);
  }
  for (int i = tid; i < 675; i += 256) sBias[i] = bias_table[i * HEADS + head];
  copy_wait();
  __syncthreads();

  // ---- QK^T: wave wv owns rows [16wv, 16wv+16) of attn, all 128 cols (8 WMMAs)
  v8f acc[8];
  {
    const v16bf a = load_A(sQ, 16 * wv, 0, HEAD_DIM);
#pragma unroll
    for (int t = 0; t < 8; ++t) {
      const v16bf b = load_BT(sK, 16 * t, 0, HEAD_DIM);
      v8f c = {0.f, 0.f, 0.f, 0.f, 0.f, 0.f, 0.f, 0.f};
      acc[t] = wmma_bf16(a, b, c);
    }
  }
  // scale + rel-pos bias + register softmax.
  // C layout: reg r, lane -> row n = rbase + r, col m = 16t + (lane&15)
  const int rbase = 16 * wv + ((lane >> 4) << 3);
  const int cm = lane & 15;
  // rbase % 8 == 0 and r < 8  =>  tn, hn invariant; wn == r
  const int tn = rbase >> 6, hn = (rbase >> 3) & 7;
  float mx[8], sm[8];
#pragma unroll
  for (int r = 0; r < 8; ++r) mx[r] = -3.0e38f;
#pragma unroll
  for (int t = 0; t < 8; ++t) {
    const int m = 16 * t + cm;
    const int tm = m >> 6, hm = (m >> 3) & 7, wm = m & 7;
    const int idx0 = (tn - tm + 1) * 225 + (hn - hm + 7) * 15 + (7 - wm);
#pragma unroll
    for (int r = 0; r < 8; ++r) {
      float val = acc[t][r] * QK_SCALE + sBias[idx0 + r];
      acc[t][r] = val;
      mx[r] = fmaxf(mx[r], val);
    }
  }
#pragma unroll
  for (int r = 0; r < 8; ++r) {
#pragma unroll
    for (int d = 8; d >= 1; d >>= 1)
      mx[r] = fmaxf(mx[r], __shfl_xor(mx[r], d, 32));  // within 16-lane half
    sm[r] = 0.f;
  }
#pragma unroll
  for (int t = 0; t < 8; ++t)
#pragma unroll
    for (int r = 0; r < 8; ++r) {
      float e = __expf(acc[t][r] - mx[r]);
      acc[t][r] = e;
      sm[r] += e;
    }
#pragma unroll
  for (int r = 0; r < 8; ++r) {
#pragma unroll
    for (int d = 8; d >= 1; d >>= 1) sm[r] += __shfl_xor(sm[r], d, 32);
    sm[r] = 1.f / sm[r];  // applied after AV (deferred normalization)
  }
  // unnormalized exp -> LDS as bf16 (row-major [128][128])
#pragma unroll
  for (int t = 0; t < 8; ++t)
#pragma unroll
    for (int r = 0; r < 8; ++r)
      sP[(rbase + r) * WIN_TOKENS + 16 * t + cm] = f2bf(acc[t][r]);
  __syncthreads();

  // ---- probs @ V : out strip [16, 32], K = 128 (4 WMMAs per 16-col tile)
#pragma unroll
  for (int nt = 0; nt < 2; ++nt) {
    v8f c = {0.f, 0.f, 0.f, 0.f, 0.f, 0.f, 0.f, 0.f};
#pragma unroll
    for (int ks = 0; ks < 4; ++ks) {
      v16bf a = load_A(sP, 16 * wv, 32 * ks, WIN_TOKENS);
      v16bf b = load_B(sV, 32 * ks, 16 * nt, HEAD_DIM);
      c = wmma_bf16(a, b, c);
    }
    const int col = head * HEAD_DIM + 16 * nt + cm;
#pragma unroll
    for (int r = 0; r < 8; ++r)
      ao[((long)win * WIN_TOKENS + rbase + r) * DIM + col] = f2bf(c[r] * sm[r]);
  }
}

// ---------------- Kernel 3: out-proj + residual + LN -> x (f32), xn (bf16), global layout ----
__global__ __launch_bounds__(256) void proj_ln_kernel(
    const unsigned short* __restrict__ ao, const unsigned short* __restrict__ wp,
    const float* __restrict__ proj_b, const float* __restrict__ vin,
    const float* __restrict__ g2, const float* __restrict__ b2,
    float* __restrict__ xbuf, unsigned short* __restrict__ xn) {
  __shared__ __attribute__((aligned(16))) unsigned short sX[16 * DIM];
  __shared__ float sO[16][DIM + 16];
  __shared__ float sRedA[256], sRedB[256];
  __shared__ float sMean[16], sRstd[16];

  const int tid = threadIdx.x, wv = tid >> 5, lane = tid & 31;
  const int win = blockIdx.x >> 3, tile = blockIdx.x & 7;

  const uint4* src = (const uint4*)(ao + ((long)win * WIN_TOKENS + tile * 16) * DIM);
  ((uint4*)sX)[tid] = src[tid];
  ((uint4*)sX)[tid + 256] = src[tid + 256];
  __syncthreads();

#pragma unroll
  for (int ct = 0; ct < 2; ++ct) {
    const int Cbase = (wv * 2 + ct) * 16;
    __builtin_prefetch(wp + (long)Cbase * DIM, 0, 0);
    v8f c = {0.f, 0.f, 0.f, 0.f, 0.f, 0.f, 0.f, 0.f};
#pragma unroll
    for (int ks = 0; ks < 8; ++ks) {
      v16bf a = load_A(sX, 0, 32 * ks, DIM);
      v16bf b = load_BT(wp, Cbase, 32 * ks, DIM);
      c = wmma_bf16(a, b, c);
    }
    const int col = Cbase + (lane & 15);
    const int r0 = (lane >> 4) << 3;
#pragma unroll
    for (int r = 0; r < 8; ++r) sO[r0 + r][col] = c[r] + proj_b[col];
  }
  __syncthreads();

  // residual + LN (16 threads per row)
  const int row = tid >> 4;
  const int c0 = (tid & 15) * 16;
  const int n = tile * 16 + row;
  const int wb = win & 7, hb = (win >> 3) & 7, tb = (win >> 6) & 3, bb = win >> 8;
  const int t_in = n >> 6, h_in = (n >> 3) & 7, w_in = n & 7;
  const long g = ((long)(bb * 8 + tb * 2 + t_in) * 64 + (hb * 8 + h_in)) * 64 +
                 (wb * 8 + w_in);
  const float* vrow = vin + g * DIM;
  float loc[16]; float s = 0.f, ss = 0.f;
#pragma unroll
  for (int j = 0; j < 16; ++j) {
    float x = sO[row][c0 + j] + vrow[c0 + j];
    loc[j] = x; s += x; ss += x * x;
  }
  sRedA[tid] = s; sRedB[tid] = ss;
  __syncthreads();
  if ((tid & 15) == 0) {
    float S = 0.f, SS = 0.f;
    for (int j = 0; j < 16; ++j) { S += sRedA[tid + j]; SS += sRedB[tid + j]; }
    float mean = S * (1.f / DIM);
    float var = SS * (1.f / DIM) - mean * mean;
    sMean[row] = mean;
    sRstd[row] = rsqrtf(var + 1e-5f);
  }
  __syncthreads();
  const float mean = sMean[row], rstd = sRstd[row];
  float* xrow = xbuf + g * DIM;
  unsigned short* xnrow = xn + g * DIM;
#pragma unroll
  for (int j = 0; j < 16; j += 2) {
    xrow[c0 + j] = loc[j];
    xrow[c0 + j + 1] = loc[j + 1];
    float a = (loc[j] - mean) * rstd * g2[c0 + j] + b2[c0 + j];
    float b = (loc[j + 1] - mean) * rstd * g2[c0 + j + 1] + b2[c0 + j + 1];
    *(unsigned*)(xnrow + c0 + j) = pack2bf(a, b);
  }
}

// ---------------- Kernel 4: h1 = gelu(xn @ mw1^T + mb1), bf16 ----------------
__global__ __launch_bounds__(256) void mlp1_kernel(
    const unsigned short* __restrict__ xn, const unsigned short* __restrict__ w1,
    const float* __restrict__ mb1, unsigned short* __restrict__ h1) {
  __shared__ __attribute__((aligned(16))) unsigned short sX[16 * DIM];
  const int tid = threadIdx.x, wv = tid >> 5, lane = tid & 31;
  const long row0 = (long)blockIdx.x * 16;
  const uint4* src = (const uint4*)(xn + row0 * DIM);
  ((uint4*)sX)[tid] = src[tid];
  ((uint4*)sX)[tid + 256] = src[tid + 256];
  __syncthreads();
  const int cmn = lane & 15, r0 = (lane >> 4) << 3;
#pragma unroll
  for (int ct = 0; ct < 4; ++ct) {
    const int Cbase = (wv * 4 + ct) * 16;
    __builtin_prefetch(w1 + (long)Cbase * DIM, 0, 0);
    v8f c = {0.f, 0.f, 0.f, 0.f, 0.f, 0.f, 0.f, 0.f};
#pragma unroll
    for (int ks = 0; ks < 8; ++ks) {
      v16bf a = load_A(sX, 0, 32 * ks, DIM);
      v16bf b = load_BT(w1, Cbase, 32 * ks, DIM);
      c = wmma_bf16(a, b, c);
    }
    const int col = Cbase + cmn;
    const float bias = mb1[col];
#pragma unroll
    for (int r = 0; r < 8; ++r) {
      float x = c[r] + bias;
      float ge = 0.5f * x * (1.f + erff(x * 0.70710678118654752f));  // exact gelu
      h1[(row0 + r0 + r) * HIDDEN + col] = f2bf(ge);
    }
  }
}

// ---------------- Kernel 5: out = x + h1 @ mw2^T + mb2 (f32) ----------------
__global__ __launch_bounds__(256) void mlp2_kernel(
    const unsigned short* __restrict__ h1, const unsigned short* __restrict__ w2,
    const float* __restrict__ mb2, const float* __restrict__ xbuf,
    float* __restrict__ out) {
  __shared__ __attribute__((aligned(16))) unsigned short sH[16 * HIDDEN];
  const int tid = threadIdx.x, wv = tid >> 5, lane = tid & 31;
  const long row0 = (long)blockIdx.x * 16;
  const uint4* src = (const uint4*)(h1 + row0 * HIDDEN);
#pragma unroll
  for (int j = 0; j < 4; ++j) ((uint4*)sH)[tid + 256 * j] = src[tid + 256 * j];
  __syncthreads();
  const int cmn = lane & 15, r0 = (lane >> 4) << 3;
#pragma unroll
  for (int ct = 0; ct < 2; ++ct) {
    const int Cbase = (wv * 2 + ct) * 16;
    __builtin_prefetch(w2 + (long)Cbase * HIDDEN, 0, 0);
    v8f c = {0.f, 0.f, 0.f, 0.f, 0.f, 0.f, 0.f, 0.f};
#pragma unroll
    for (int ks = 0; ks < 16; ++ks) {
      v16bf a = load_A(sH, 0, 32 * ks, HIDDEN);
      v16bf b = load_BT(w2, Cbase, 32 * ks, HIDDEN);
      c = wmma_bf16(a, b, c);
    }
    const int col = Cbase + cmn;
    const float bias = mb2[col];
#pragma unroll
    for (int r = 0; r < 8; ++r) {
      const long gi = (row0 + r0 + r) * DIM + col;
      out[gi] = c[r] + bias + xbuf[gi];
    }
  }
}

extern "C" void kernel_launch(void* const* d_in, const int* in_sizes, int n_in,
                              void* d_out, int out_size, void* d_ws, size_t ws_size,
                              hipStream_t stream) {
  const float* q = (const float*)d_in[0];
  const float* k = (const float*)d_in[1];
  const float* v = (const float*)d_in[2];
  const float* gq = (const float*)d_in[3];
  const float* bq = (const float*)d_in[4];
  const float* gk = (const float*)d_in[5];
  const float* bk = (const float*)d_in[6];
  const float* gv = (const float*)d_in[7];
  const float* bv = (const float*)d_in[8];
  const float* bias_table = (const float*)d_in[9];
  const float* proj_w = (const float*)d_in[10];
  const float* proj_b = (const float*)d_in[11];
  const float* g2 = (const float*)d_in[12];
  const float* b2 = (const float*)d_in[13];
  const float* mw1 = (const float*)d_in[14];
  const float* mb1 = (const float*)d_in[15];
  const float* mw2 = (const float*)d_in[16];
  const float* mb2 = (const float*)d_in[17];

  char* ws = (char*)d_ws;
  const size_t SZ_BF = 16777216ull * 2;  // 33.5 MB per bf16 [65536,256] buffer
  unsigned short* qh = (unsigned short*)(ws);
  unsigned short* kh = (unsigned short*)(ws + SZ_BF);
  unsigned short* vh = (unsigned short*)(ws + 2 * SZ_BF);
  unsigned short* ao = (unsigned short*)(ws + 3 * SZ_BF);
  float* xbuf = (float*)(ws + 4 * SZ_BF);  // 67 MB f32
  unsigned short* xn = (unsigned short*)(ws + 4 * SZ_BF + 16777216ull * 4);
  unsigned short* wbf = (unsigned short*)(ws + 5 * SZ_BF + 16777216ull * 4);
  unsigned short* wpbf = wbf;            // proj_w bf16 [256,256]
  unsigned short* w1bf = wbf + 65536;    // mw1 bf16 [512,256]
  unsigned short* w2bf = wbf + 196608;   // mw2 bf16 [256,512]
  unsigned short* h1 = (unsigned short*)(ws);  // reuse qh+kh (dead after attention)

  cvt_w_kernel<<<160, 256, 0, stream>>>(proj_w, mw1, mw2, wbf);
  ln_qkv_kernel<<<24576, 256, 0, stream>>>(q, k, v, gq, bq, gk, bk, gv, bv,
                                           qh, kh, vh);
  win_attn_kernel<<<NWIN * HEADS, 256, 0, stream>>>(qh, kh, vh, bias_table, ao);
  proj_ln_kernel<<<NWIN * 8, 256, 0, stream>>>(ao, wpbf, proj_b, v, g2, b2,
                                               xbuf, xn);
  mlp1_kernel<<<4096, 256, 0, stream>>>(xn, w1bf, mb1, h1);
  mlp2_kernel<<<4096, 256, 0, stream>>>(h1, w2bf, mb2, xbuf, (float*)d_out);
}
